// RWKV_BLOCK_4827543240832
// MI455X (gfx1250) — matrix operand order repaired and checked by reading, not compiled, not run on hardware
//
#include <hip/hip_runtime.h>
#include <hip/hip_bf16.h>
#include <math.h>

#define BB 512
#define EE 2048
#define HH 32
#define SS 64
#define RB 8   // batch rows per LoRA block

typedef __attribute__((ext_vector_type(16))) __bf16 v16bf;
typedef __attribute__((ext_vector_type(8)))  __bf16 v8bf;
typedef __attribute__((ext_vector_type(8)))  float  v8f;
typedef __attribute__((ext_vector_type(4)))  float  v4f;
typedef __attribute__((ext_vector_type(4)))  int    v4i;

static __device__ __forceinline__ float sigm(float x){ return 1.0f/(1.0f+expf(-x)); }

static __device__ __forceinline__ float blk256_sum(float v, float* red){
  int tid = threadIdx.x;
  red[tid]=v; __syncthreads();
  for (int o=128;o>0;o>>=1){ if(tid<o) red[tid]+=red[tid+o]; __syncthreads(); }
  float r = red[0]; __syncthreads();
  return r;
}
static __device__ __forceinline__ float blk64_sum(float v, float* red){
  int tid = threadIdx.x;
  red[tid]=v; __syncthreads();
  for (int o=32;o>0;o>>=1){ if(tid<o) red[tid]+=red[tid+o]; __syncthreads(); }
  float r = red[0]; __syncthreads();
  return r;
}

// ---- async global->LDS 16B copy (ASYNCcnt path), with safe fallback ----
static __device__ __forceinline__ void async_cp16(const __bf16* g, __bf16* l){
#if __has_builtin(__builtin_amdgcn_global_load_async_to_lds_b128)
  __builtin_amdgcn_global_load_async_to_lds_b128(
      (__attribute__((address_space(1))) v4i*)g,
      (__attribute__((address_space(3))) v4i*)l, 0, 0);
#else
  *(v8bf*)l = *(const v8bf*)g;
#endif
}
static __device__ __forceinline__ void wait_async_le8(){
#if __has_builtin(__builtin_amdgcn_global_load_async_to_lds_b128)
  asm volatile("s_wait_asynccnt 0x8" ::: "memory");
#endif
}
static __device__ __forceinline__ void wait_async_0(){
#if __has_builtin(__builtin_amdgcn_global_load_async_to_lds_b128)
  asm volatile("s_wait_asynccnt 0x0" ::: "memory");
#endif
}

// ---------------- fp32 -> bf16 conversion (8 elems/thread) ----------------
__global__ __launch_bounds__(256)
void k_cvt_bf16(const float* __restrict__ s, __bf16* __restrict__ d, int n){
  int i = (blockIdx.x*256 + threadIdx.x)*8;
  if (i < n){
    v4f a = *(const v4f*)(s+i);
    v4f b = *(const v4f*)(s+i+4);
    v8bf o;
    o[0]=(__bf16)a.x; o[1]=(__bf16)a.y; o[2]=(__bf16)a.z; o[3]=(__bf16)a.w;
    o[4]=(__bf16)b.x; o[5]=(__bf16)b.y; o[6]=(__bf16)b.z; o[7]=(__bf16)b.w;
    *(v8bf*)(d+i) = o;
  }
}

// ---------------- LayerNorm + 6-way token-shift mix ----------------
__global__ __launch_bounds__(256)
void k_ln_mix6(const float* __restrict__ x, const float* __restrict__ st1,
               const float* __restrict__ lw, const float* __restrict__ lb,
               const float* __restrict__ xmix, float* __restrict__ mixed){
  __shared__ float red[256];
  int b = blockIdx.x, tid = threadIdx.x;
  const float* xr = x + (size_t)b*EE;
  float s=0.f, ss=0.f;
  for (int e=tid;e<EE;e+=256){ float v=xr[e]; s+=v; ss+=v*v; }
  float mu  = blk256_sum(s, red)*(1.0f/EE);
  float ex2 = blk256_sum(ss,red)*(1.0f/EE);
  float rs  = rsqrtf(ex2 - mu*mu + 1e-5f);
  const float* s1 = st1 + (size_t)b*EE;
  const size_t BE = (size_t)BB*EE;
  for (int e=tid;e<EE;e+=256){
    float xl = (xr[e]-mu)*rs*lw[e]+lb[e];
    float sx = s1[e]-xl;
    #pragma unroll
    for (int j=0;j<6;j++)
      mixed[(size_t)j*BE + (size_t)b*EE + e] = xl + sx*xmix[j*EE+e];
  }
}

__global__ __launch_bounds__(256)
void k_ln_mix1(const float* __restrict__ x, const float* __restrict__ stc,
               const float* __restrict__ lw, const float* __restrict__ lb,
               const float* __restrict__ fxk, float* __restrict__ xkf){
  __shared__ float red[256];
  int b = blockIdx.x, tid = threadIdx.x;
  const float* xr = x + (size_t)b*EE;
  float s=0.f, ss=0.f;
  for (int e=tid;e<EE;e+=256){ float v=xr[e]; s+=v; ss+=v*v; }
  float mu  = blk256_sum(s, red)*(1.0f/EE);
  float ex2 = blk256_sum(ss,red)*(1.0f/EE);
  float rs  = rsqrtf(ex2 - mu*mu + 1e-5f);
  const float* sc = stc + (size_t)b*EE;
  for (int e=tid;e<EE;e+=256){
    float xl = (xr[e]-mu)*rs*lw[e]+lb[e];
    xkf[(size_t)b*EE + e] = xl + (sc[e]-xl)*fxk[e];
  }
}

// ---------------- Fused LoRA (w/a/v/g), RB rows per block ----------------
__global__ __launch_bounds__(256)
void k_lora(const float* __restrict__ mixed,
            const float* __restrict__ w0, const float* __restrict__ w1, const float* __restrict__ w2,
            const float* __restrict__ a0, const float* __restrict__ a1, const float* __restrict__ a2,
            const float* __restrict__ v0, const float* __restrict__ v1, const float* __restrict__ v2,
            const float* __restrict__ g1, const float* __restrict__ g2,
            float* __restrict__ decay, float* __restrict__ a_s,
            float* __restrict__ vgate, float* __restrict__ g_o){
  __shared__ float xl[RB][EE];     // 64KB: RB activation rows
  __shared__ float t[RB][128];     // stage-1 results
  int b0 = blockIdx.x*RB, tid = threadIdx.x;
  const size_t BE = (size_t)BB*EE;

  auto stage1 = [&](const float* xin, const float* W1, int D, int act1){
    for (int i=tid; i<RB*EE; i+=256){
      int row = i >> 11, e = i & (EE-1);
      xl[row][e] = xin[(size_t)(b0+row)*EE + e];
    }
    __syncthreads();
    for (int p = tid; p < RB*D; p += 256){
      int row = p / D, d = p % D;
      float s = 0.f;
      for (int k=0;k<EE;k++) s += xl[row][k]*W1[(size_t)k*D+d];
      t[row][d] = (act1==1)?tanhf(s):((act1==2)?sigm(s):s);
    }
    __syncthreads();
  };

  // w: decay = exp(-0.606531*sigmoid(w0 + tanh(xw@w1)@w2))
  stage1(mixed + 1*BE, w1, 64, 1);
  for (int e=tid;e<EE;e+=256){
    float acc[RB];
    #pragma unroll
    for (int r2=0;r2<RB;r2++) acc[r2]=0.f;
    for (int d=0; d<64; d++){
      float wv = w2[(size_t)d*EE+e];
      #pragma unroll
      for (int r2=0;r2<RB;r2++) acc[r2] += t[r2][d]*wv;
    }
    float bz = w0[e];
    #pragma unroll
    for (int r2=0;r2<RB;r2++)
      decay[(size_t)(b0+r2)*EE+e] = expf(-0.606531f * sigm(bz+acc[r2]));
  }
  __syncthreads();
  // a: sigmoid(a0 + xa@a1@a2)
  stage1(mixed + 4*BE, a1, 64, 0);
  for (int e=tid;e<EE;e+=256){
    float acc[RB];
    #pragma unroll
    for (int r2=0;r2<RB;r2++) acc[r2]=0.f;
    for (int d=0; d<64; d++){
      float wv = a2[(size_t)d*EE+e];
      #pragma unroll
      for (int r2=0;r2<RB;r2++) acc[r2] += t[r2][d]*wv;
    }
    float bz = a0[e];
    #pragma unroll
    for (int r2=0;r2<RB;r2++)
      a_s[(size_t)(b0+r2)*EE+e] = sigm(bz+acc[r2]);
  }
  __syncthreads();
  // v gate: sigmoid(v0 + xv@v1@v2)
  stage1(mixed + 3*BE, v1, 32, 0);
  for (int e=tid;e<EE;e+=256){
    float acc[RB];
    #pragma unroll
    for (int r2=0;r2<RB;r2++) acc[r2]=0.f;
    for (int d=0; d<32; d++){
      float wv = v2[(size_t)d*EE+e];
      #pragma unroll
      for (int r2=0;r2<RB;r2++) acc[r2] += t[r2][d]*wv;
    }
    float bz = v0[e];
    #pragma unroll
    for (int r2=0;r2<RB;r2++)
      vgate[(size_t)(b0+r2)*EE+e] = sigm(bz+acc[r2]);
  }
  __syncthreads();
  // g: sigmoid(xg@g1)@g2
  stage1(mixed + 5*BE, g1, 128, 2);
  for (int e=tid;e<EE;e+=256){
    float acc[RB];
    #pragma unroll
    for (int r2=0;r2<RB;r2++) acc[r2]=0.f;
    for (int d=0; d<128; d++){
      float wv = g2[(size_t)d*EE+e];
      #pragma unroll
      for (int r2=0;r2<RB;r2++) acc[r2] += t[r2][d]*wv;
    }
    #pragma unroll
    for (int r2=0;r2<RB;r2++)
      g_o[(size_t)(b0+r2)*EE+e] = acc[r2];
  }
}

// ---- bf16 WMMA GEMM, async double-buffered: C[M,N] = epi(A[M,K] @ W[N,K]^T) ----
// EPI: 0=none, 1=+ADD, 2=relu^2
template<int EPI>
__global__ __launch_bounds__(128)
void k_gemm_bf16(const __bf16* __restrict__ A, const __bf16* __restrict__ W,
                 const float* __restrict__ ADD, float* __restrict__ C,
                 int M, int N, int K){
  __shared__ __bf16 As[2][64][72];
  __shared__ __bf16 Bs[2][64][72];
  int tid = threadIdx.x;
  int lane = tid & 31, wv = tid >> 5;
  int wm = wv >> 1, wn = wv & 1;           // 2x2 wave grid, 32x32 per wave
  int bm = blockIdx.y, bn = blockIdx.x;
  int lrow = lane & 15, lhi = lane >> 4;
  v8f acc[2][2] = {};

  // per-thread copy coords: 512 16B-chunks per 64x64 bf16 tile, 4 per thread
  const __bf16* gA[4]; const __bf16* gB[4];
  int rT[4], cT[4];
  #pragma unroll
  for (int i=0;i<4;i++){
    int id = tid + i*128;
    int r = id >> 3, c = id & 7;
    rT[i]=r; cT[i]=c;
    gA[i] = A + (size_t)(bm*64 + r)*K + c*8;
    gB[i] = W + (size_t)(bn*64 + r)*K + c*8;
  }
  auto issue = [&](int buf, int k0){
    #pragma unroll
    for (int i=0;i<4;i++){
      async_cp16(gA[i] + k0, &As[buf][rT[i]][cT[i]*8]);
      async_cp16(gB[i] + k0, &Bs[buf][rT[i]][cT[i]*8]);
    }
  };

  issue(0, 0);
  const int T = K >> 6;
  for (int t = 0; t < T; t++){
    if (t+1 < T){ issue((t+1)&1, (t+1)<<6); wait_async_le8(); }
    else        { wait_async_0(); }
    __syncthreads();                 // tile t visible to all waves
    int buf = t & 1;
    #pragma unroll
    for (int kk = 0; kk < 64; kk += 32){
      v16bf afr[2], bfr[2];
      #pragma unroll
      for (int mt=0; mt<2; mt++){
        int m  = wm*32 + mt*16 + lrow;
        int kh = kk + lhi*8;         // lanes 0-15: K 0..7 & 16..23; lanes 16-31: K 8..15 & 24..31
        union { v16bf v; v8bf h[2]; } u;
        u.h[0] = *(const v8bf*)&As[buf][m][kh];
        u.h[1] = *(const v8bf*)&As[buf][m][kh+16];
        afr[mt] = u.v;
      }
      #pragma unroll
      for (int nt=0; nt<2; nt++){
        int n  = wn*32 + nt*16 + lrow;
        int kb = kk + lhi*8;         // lanes 0-15: K 0..7 & 16..23; lanes 16-31: K 8..15 & 24..31
        union { v16bf v; v8bf h[2]; } u;
        u.h[0] = *(const v8bf*)&Bs[buf][n][kb];
        u.h[1] = *(const v8bf*)&Bs[buf][n][kb+16];
        bfr[nt] = u.v;
      }
      #pragma unroll
      for (int mt=0; mt<2; mt++)
        #pragma unroll
        for (int nt=0; nt<2; nt++)
          acc[mt][nt] = __builtin_amdgcn_wmma_f32_16x16x32_bf16(
              false, afr[mt], false, bfr[nt], (short)0, acc[mt][nt], false, false);
    }
    __syncthreads();                 // all waves done reading buf before overwrite
  }

  #pragma unroll
  for (int mt=0; mt<2; mt++)
    #pragma unroll
    for (int nt=0; nt<2; nt++){
      int n = bn*64 + wn*32 + nt*16 + lrow;
      #pragma unroll
      for (int vg=0; vg<8; vg++){
        int m = bm*64 + wm*32 + mt*16 + vg + lhi*8;
        float v = acc[mt][nt][vg];
        size_t o = (size_t)m*N + n;
        if (EPI==1) v += ADD[o];
        if (EPI==2) { float t2 = v>0.f?v:0.f; v = t2*t2; }
        C[o] = v;
      }
    }
}

// ---------------- per-head: normalize kk, k2, v-lerp ----------------
__global__ __launch_bounds__(64)
void k_head(const float* __restrict__ k_arr, const float* __restrict__ a_s,
            const float* vraw, const float* __restrict__ vfirst,
            const float* __restrict__ vgate,
            const float* __restrict__ k_k, const float* __restrict__ k_a,
            float* __restrict__ kkn, float* __restrict__ kka,
            float* __restrict__ k2, float* vout){
  __shared__ float red[64];
  int h = blockIdx.x, b = blockIdx.y, j = threadIdx.x;
  int e = h*SS + j;
  size_t be = (size_t)b*EE + e;
  float kv  = k_arr[be];
  float kkv = kv * k_k[e];
  float ssq = blk64_sum(kkv*kkv, red);
  float inv = 1.0f / fmaxf(sqrtf(ssq), 1e-12f);
  float kknv = kkv*inv;
  float av = a_s[be];
  kkn[be] = kknv;
  kka[be] = kknv*av;
  k2[be]  = kv*(1.0f + (av-1.0f)*k_a[e]);
  float vr = vraw[be];
  vout[be] = vr + (vfirst[be]-vr)*vgate[be];
}

// -------- state update + out (rank-1 delta rule -> two fused matvecs) --------
__global__ __launch_bounds__(64)
void k_state(const float* __restrict__ st2, const float* __restrict__ r,
             const float* __restrict__ decay, const float* __restrict__ kkn,
             const float* __restrict__ kka, const float* __restrict__ k2,
             const float* __restrict__ v, float* __restrict__ outs){
  __shared__ float sl[64][65];
  __shared__ float wr[64], kn[64], red[64];
  int h = blockIdx.x, b = blockIdx.y, j = threadIdx.x;
  size_t base = (size_t)b*EE + (size_t)h*SS;
  float rj = r[base+j];
  wr[j] = decay[base+j]*rj;
  kn[j] = kkn[base+j];
  const float* sp = st2 + ((size_t)(b*HH + h))*SS*SS;
  for (int i=0;i<64;i++) sl[i][j] = sp[(size_t)i*64 + j];
  float c2 = blk64_sum(kka[base+j]*rj, red);   // kka . r
  float c3 = blk64_sum(k2[base+j]*rj, red);    // k2  . r
  float acc1=0.f, acc2=0.f;
  for (int k=0;k<64;k++){ float sv = sl[j][k]; acc1 += sv*wr[k]; acc2 += sv*kn[k]; }
  outs[base+j] = acc1 - acc2*c2 + v[base+j]*c3;
}

// ---------------- GroupNorm + rkv + g-gate ----------------
__global__ __launch_bounds__(64)
void k_gn(const float* __restrict__ outs, const float* __restrict__ r,
          const float* __restrict__ k2, const float* __restrict__ v,
          const float* __restrict__ g_o, const float* __restrict__ r_k,
          const float* __restrict__ gnw, const float* __restrict__ gnb,
          float* __restrict__ attin){
  __shared__ float red[64];
  int h = blockIdx.x, b = blockIdx.y, j = threadIdx.x;
  size_t base = (size_t)b*EE + (size_t)h*SS;
  int e = h*SS + j;
  float o  = outs[base+j];
  float mu = blk64_sum(o, red)*(1.0f/SS);
  float var= blk64_sum((o-mu)*(o-mu), red)*(1.0f/SS);
  float gn = (o-mu)*rsqrtf(var + 0.00064f)*gnw[e] + gnb[e];
  float rj = r[base+j];
  float rdot = blk64_sum(rj*k2[base+j]*r_k[e], red);
  attin[base+j] = (gn + rdot*v[base+j]) * g_o[base+j];
}

__global__ void k_copy(const float* __restrict__ a, float* __restrict__ o, int n){
  int i = blockIdx.x*256 + threadIdx.x;
  if (i < n) o[i] = a[i];
}

extern "C" void kernel_launch(void* const* d_in, const int* in_sizes, int n_in,
                              void* d_out, int out_size, void* d_ws, size_t ws_size,
                              hipStream_t stream){
  const float* x     = (const float*)d_in[0];
  const float* vfirst= (const float*)d_in[1];
  const float* stc   = (const float*)d_in[2];
  const float* st1   = (const float*)d_in[3];
  const float* st2   = (const float*)d_in[4];
  const float* ln1w  = (const float*)d_in[5];
  const float* ln1b  = (const float*)d_in[6];
  const float* ln2w  = (const float*)d_in[7];
  const float* ln2b  = (const float*)d_in[8];
  const float* xmix  = (const float*)d_in[9];
  const float* w0 = (const float*)d_in[10];
  const float* w1 = (const float*)d_in[11];
  const float* w2 = (const float*)d_in[12];
  const float* a0 = (const float*)d_in[13];
  const float* a1 = (const float*)d_in[14];
  const float* a2 = (const float*)d_in[15];
  const float* v0 = (const float*)d_in[16];
  const float* v1 = (const float*)d_in[17];
  const float* v2 = (const float*)d_in[18];
  const float* g1 = (const float*)d_in[19];
  const float* g2 = (const float*)d_in[20];
  const float* k_k = (const float*)d_in[21];
  const float* k_a = (const float*)d_in[22];
  const float* r_k = (const float*)d_in[23];
  const float* Wr  = (const float*)d_in[24];
  const float* Wk  = (const float*)d_in[25];
  const float* Wv  = (const float*)d_in[26];
  const float* Wo  = (const float*)d_in[27];
  const float* gnw = (const float*)d_in[28];
  const float* gnb = (const float*)d_in[29];
  const float* fxk = (const float*)d_in[30];
  const float* Wkf = (const float*)d_in[31];
  const float* Wvf = (const float*)d_in[32];

  float* ws = (float*)d_ws;
  const size_t BE = (size_t)BB*EE;
  const size_t EEE = (size_t)EE*EE;
  float* mixed = ws;             // [6][B][E]: xr,xw,xk,xv,xa,xg
  float* decay = ws + 6*BE;
  float* a_s   = ws + 7*BE;
  float* vgate = ws + 8*BE;
  float* g_o   = ws + 9*BE;
  float* r_b   = ws + 10*BE;
  float* k_b   = ws + 11*BE;
  float* v_b   = ws + 12*BE;
  float* x1    = ws + 13*BE;
  __bf16* abf  = (__bf16*)(ws + 14*BE);   // BE bf16 (activation staging)
  __bf16* wbf  = abf + BE;                // E*E bf16 (weight staging)
  // slot reuse after their consumers finish (stream-ordered):
  float* kkn   = mixed + 1*BE;   // was xw
  float* kka   = mixed + 4*BE;   // was xa
  float* k2b   = mixed + 5*BE;   // was xg
  float* outs  = mixed + 0*BE;   // was xr
  float* attin = mixed + 2*BE;   // was xk
  float* xkf   = mixed + 3*BE;   // was xv
  float* kf    = r_b;            // r consumed by k_state/k_gn

  float* xout = (float*)d_out;        // x2
  float* vout = (float*)d_out + BE;   // v_first passthrough

  dim3 gg(EE/64, BB/64);
  dim3 bh(HH, BB);
  const int cvA = (int)(BE/8/256);    // grids for conversions (exact multiples)
  const int cvW = (int)(EEE/8/256);

  k_ln_mix6<<<BB,256,0,stream>>>(x, st1, ln1w, ln1b, xmix, mixed);
  k_lora<<<BB/RB,256,0,stream>>>(mixed, w0,w1,w2, a0,a1,a2, v0,v1,v2, g1,g2,
                                 decay, a_s, vgate, g_o);

  k_cvt_bf16<<<cvA,256,0,stream>>>(mixed+0*BE, abf, (int)BE);
  k_cvt_bf16<<<cvW,256,0,stream>>>(Wr, wbf, (int)EEE);
  k_gemm_bf16<0><<<gg,128,0,stream>>>(abf, wbf, nullptr, r_b, BB, EE, EE);

  k_cvt_bf16<<<cvA,256,0,stream>>>(mixed+2*BE, abf, (int)BE);
  k_cvt_bf16<<<cvW,256,0,stream>>>(Wk, wbf, (int)EEE);
  k_gemm_bf16<0><<<gg,128,0,stream>>>(abf, wbf, nullptr, k_b, BB, EE, EE);

  k_cvt_bf16<<<cvA,256,0,stream>>>(mixed+3*BE, abf, (int)BE);
  k_cvt_bf16<<<cvW,256,0,stream>>>(Wv, wbf, (int)EEE);
  k_gemm_bf16<0><<<gg,128,0,stream>>>(abf, wbf, nullptr, v_b, BB, EE, EE);

  k_head<<<bh,64,0,stream>>>(k_b, a_s, v_b, vfirst, vgate, k_k, k_a,
                             kkn, kka, k2b, v_b);
  k_state<<<bh,64,0,stream>>>(st2, r_b, decay, kkn, kka, k2b, v_b, outs);
  k_gn<<<bh,64,0,stream>>>(outs, r_b, k2b, v_b, g_o, r_k, gnw, gnb, attin);

  k_cvt_bf16<<<cvA,256,0,stream>>>(attin, abf, (int)BE);
  k_cvt_bf16<<<cvW,256,0,stream>>>(Wo, wbf, (int)EEE);
  k_gemm_bf16<1><<<gg,128,0,stream>>>(abf, wbf, x, x1, BB, EE, EE);

  k_ln_mix1<<<BB,256,0,stream>>>(x1, stc, ln2w, ln2b, fxk, xkf);

  k_cvt_bf16<<<cvA,256,0,stream>>>(xkf, abf, (int)BE);
  k_cvt_bf16<<<cvW,256,0,stream>>>(Wkf, wbf, (int)EEE);
  k_gemm_bf16<2><<<gg,128,0,stream>>>(abf, wbf, nullptr, kf, BB, EE, EE);

  k_cvt_bf16<<<cvA,256,0,stream>>>(kf, abf, (int)BE);
  k_cvt_bf16<<<cvW,256,0,stream>>>(Wvf, wbf, (int)EEE);
  k_gemm_bf16<1><<<gg,128,0,stream>>>(abf, wbf, x1, xout, BB, EE, EE);

  k_copy<<<(int)((BE+255)/256),256,0,stream>>>(vfirst, vout, (int)BE);
}
